// ROIGather_74354473828521
// MI455X (gfx1250) — compile-verified, matrix operand order checked by reading, not run on hardware
//
#include <hip/hip_runtime.h>

// ---------------------------------------------------------------------------
// ROIGather for MI455X (gfx1250): f16 WMMA (16x16x32, f32 accum) for all GEMM-
// shaped work; BN folded to scale/shift; fkey∘a3 pre-collapsed; attention key/
// value computed only at the 250 resize-sampled positions. Weight slices are
// staged into LDS with GLOBAL_LOAD_ASYNC_TO_LDS_B128 (ASYNCcnt) and prefetched.
// ---------------------------------------------------------------------------

typedef __attribute__((ext_vector_type(16))) _Float16 v16h;
typedef __attribute__((ext_vector_type(8)))  _Float16 v8h;
typedef __attribute__((ext_vector_type(8)))  float    v8f;

union Frag { v16h v; v8h h[2]; };

#define WMMA16(a, b, c) \
  __builtin_amdgcn_wmma_f32_16x16x32_f16(false, (a), false, (b), (short)0, (c), false, false)

// Async 16B global->LDS copy. LDS operand = AS(3) byte offset (low 32 bits of
// the flat shared pointer, per the CDNA5 LDS-aperture truncation rule).
__device__ __forceinline__ void async_copy16(const void* lds_ptr, const void* gptr) {
  unsigned lds = (unsigned)(size_t)lds_ptr;
  unsigned long long ga = (unsigned long long)(size_t)gptr;
  asm volatile("global_load_async_to_lds_b128 %0, %1, off"
               :: "v"(lds), "v"(ga) : "memory");
}
__device__ __forceinline__ void async_wait0() {
  asm volatile("s_wait_asynccnt 0x0" ::: "memory");
}

// Load a 16x32 f16 fragment from row-major storage (row dim -> lanes, K -> regs).
// rowBase must already include the K-chunk column offset. stride in halves
// (multiple of 8 so each lane's 16B loads stay aligned).
__device__ __forceinline__ v16h frag_ld(const _Float16* rowBase, int stride) {
  const int lane = threadIdx.x & 31;
  const _Float16* p = rowBase + (lane & 15) * stride + ((lane >> 4) << 3);
  Frag f;
  f.h[0] = *(const v8h*)(p);
  f.h[1] = *(const v8h*)(p + 16);
  return f.v;
}

// ---------------- workspace layout (bytes) ----------------
__host__ __device__ constexpr size_t alignup(size_t x) { return (x + 255) & ~size_t(255); }
constexpr size_t OFF_W1H   = 0;                                   // [l][t][o][c] 3*9*48*64 f16
constexpr size_t OFF_W2H   = alignup(OFF_W1H + 3*9*48*64*2);      // [t][o][c]    9*64*144 f16
constexpr size_t OFF_FCH   = alignup(OFF_W2H + (size_t)9*64*144*2);      // 64*2304 f16
constexpr size_t OFF_FVH   = alignup(OFF_FCH + (size_t)64*2304*2);       // 64*64 f16
constexpr size_t OFF_FKA3H = alignup(OFF_FVH + 64*64*2);          // 64*64 f16
constexpr size_t OFF_FKA3B = alignup(OFF_FKA3H + 64*64*2);        // 64 f32
constexpr size_t OFF_S1    = alignup(OFF_FKA3B + 64*4);           // 144 f32
constexpr size_t OFF_H1    = alignup(OFF_S1 + 144*4);
constexpr size_t OFF_S2    = alignup(OFF_H1 + 144*4);             // 64 f32
constexpr size_t OFF_H2    = alignup(OFF_S2 + 64*4);
constexpr size_t OFF_SK    = alignup(OFF_H2 + 64*4);
constexpr size_t OFF_HK    = alignup(OFF_SK + 64*4);
constexpr size_t OFF_ROI   = alignup(OFF_HK + 64*4);              // 3072*2304 f16
constexpr size_t OFF_RWS   = alignup(OFF_ROI + (size_t)3072*2304*2);  // 3072*64 f32
constexpr size_t OFF_QWS   = alignup(OFF_RWS + (size_t)3072*64*4);    // 3072*64 f16
constexpr size_t OFF_ATT   = alignup(OFF_QWS + (size_t)3072*64*2);    // 16*256*64 f16
constexpr size_t OFF_XSM   = alignup(OFF_ATT + (size_t)16*256*64*2);  // 16*256*64 f16
constexpr size_t OFF_KWS   = alignup(OFF_XSM + (size_t)16*256*64*2);  // 16*256*64 f16 ([l][c])
constexpr size_t OFF_VT    = alignup(OFF_KWS + (size_t)16*256*64*2);  // 16*64*256 f16 ([c][l])

// ======================== prep kernels ========================
__global__ void prep_w1(const float* __restrict__ w, _Float16* __restrict__ out) {
  int i = blockIdx.x * 256 + threadIdx.x;
  if (i >= 3*9*48*64) return;
  int c = i & 63, r = i >> 6;
  int o = r % 48; r /= 48;
  int t = r % 9;  int l = r / 9;
  out[i] = (_Float16)w[((size_t)(l*48 + o)*64 + c)*9 + t];
}

__global__ void prep_w2(const float* __restrict__ w, _Float16* __restrict__ out) {
  int i = blockIdx.x * 256 + threadIdx.x;
  if (i >= 9*64*144) return;
  int c = i % 144, r = i / 144;
  int o = r & 63, t = r >> 6;
  out[i] = (_Float16)w[((size_t)o*144 + c)*9 + t];
}

__global__ void prep_cast_f16(const float* __restrict__ in, _Float16* __restrict__ out, int n) {
  int i = blockIdx.x * blockDim.x + threadIdx.x;
  if (i < n) out[i] = (_Float16)in[i];
}

__global__ void prep_fka3(const float* __restrict__ fk, const float* __restrict__ a3,
                          const float* __restrict__ a3b,
                          _Float16* __restrict__ outm, float* __restrict__ outb) {
  int i = blockIdx.x * 256 + threadIdx.x;          // 0..4095
  int o = i >> 6, j = i & 63;
  float acc = 0.f;
  for (int m = 0; m < 64; ++m) acc += fk[o*64 + m] * a3[m*64 + j];
  outm[i] = (_Float16)acc;
  if (blockIdx.x == 0 && threadIdx.x < 64) {
    float b = 0.f;
    for (int m = 0; m < 64; ++m) b += fk[threadIdx.x*64 + m] * a3b[m];
    outb[threadIdx.x] = b;
  }
}

__global__ void prep_bn(const float* cg, const float* cb, const float* cm, const float* cv,
                        const float* g2, const float* b2, const float* m2, const float* v2,
                        const float* gk, const float* bk, const float* mk, const float* vk,
                        float* S1, float* H1, float* S2, float* H2, float* SK, float* HK) {
  int t = threadIdx.x;
  if (t < 144)      { float s = cg[t]*rsqrtf(cv[t]+1e-5f); S1[t]=s; H1[t]=cb[t]-cm[t]*s; }
  else if (t < 208) { int j=t-144; float s = g2[j]*rsqrtf(v2[j]+1e-5f); S2[j]=s; H2[j]=b2[j]-m2[j]*s; }
  else if (t < 272) { int j=t-208; float s = gk[j]*rsqrtf(vk[j]+1e-5f); SK[j]=s; HK[j]=bk[j]-mk[j]*s; }
}

// ======================== kernel A: per-ROI fused convs ========================
// conv1 (x3 layers): out[s][o] = sum_t sum_c W[l][t][o][c] * in[c][s+t-4]
// per tap: GEMM 48x48, K=64. catconv: GEMM 48x64, K=144(pad 160) per tap.
__global__ __launch_bounds__(256) void roi_conv_kernel(
    const float* __restrict__ rf, const _Float16* __restrict__ W1H,
    const _Float16* __restrict__ W2H, const float* __restrict__ S1,
    const float* __restrict__ H1, const float* __restrict__ S2,
    const float* __restrict__ H2, _Float16* __restrict__ roi_out) {
  constexpr int XS  = 72;    // xin row stride (halves)
  constexpr int FS  = 168;   // fpad / w2 row stride
  constexpr int W1S = 72;
  __shared__ __align__(16) _Float16 sm_xin[3*56*XS];   // 24192 B; aliased as w2 in phase 2
  __shared__ __align__(16) _Float16 sm_fpad[56*FS];    // 18816 B
  __shared__ __align__(16) _Float16 sm_w1[48*W1S];     //  6912 B

  const int n = blockIdx.x;
  const int tid = threadIdx.x;
  const int wave = __builtin_amdgcn_readfirstlane(tid >> 5);   // scalar wave id
  const int lane = tid & 31;
  const int col = lane & 15, rbase = (lane >> 4) * 8;

  for (int i = tid; i < 3*56*XS; i += 256) sm_xin[i] = (_Float16)0.f;
  for (int i = tid; i < 56*FS;   i += 256) sm_fpad[i] = (_Float16)0.f;
  __syncthreads();
  // load ROI input -> [l][s+4][c], f16 (transpose+convert: stays synchronous)
  for (int i = tid; i < 3*64*36; i += 256) {
    int l = i / 2304, rem = i - l*2304;
    int c = rem / 36, s = rem - c*36;
    float v = rf[((size_t)(l*3072 + n)*64 + c)*36 + s];
    sm_xin[l*56*XS + (s+4)*XS + c] = (_Float16)v;
  }
  __syncthreads();

  // ---- phase 1: three per-layer convs -> sm_fpad (channel blocks of 48) ----
  for (int l = 0; l < 3; ++l) {
    v8f z = {};
    v8f acc[2]; acc[0] = z; acc[1] = z;
    for (int t = 0; t < 9; ++t) {
      __syncthreads();
      const _Float16* wsrc = W1H + (size_t)(l*9 + t)*48*64;
      // async-stage 48x64 f16 weight slice: 384 x 16B chunks (wave-aligned tail)
      for (int ch = tid; ch < 384; ch += 256) {
        int o = ch >> 3, part = ch & 7;
        async_copy16(&sm_w1[o*W1S + part*8], wsrc + o*64 + part*8);
      }
      if (t < 8) __builtin_prefetch(wsrc + 48*64, 0, 1);   // next tap slice -> L2
      async_wait0();
      __syncthreads();
      for (int tt = wave, slot = 0; tt < 9; tt += 8, ++slot) {
        int mt = tt / 3, nt = tt - mt*3;
        for (int kc = 0; kc < 2; ++kc) {
          v16h a = frag_ld(sm_xin + l*56*XS + (mt*16 + t)*XS + kc*32, XS);
          v16h b = frag_ld(sm_w1 + nt*16*W1S + kc*32, W1S);
          acc[slot] = WMMA16(a, b, acc[slot]);
        }
      }
    }
    for (int tt = wave, slot = 0; tt < 9; tt += 8, ++slot) {
      int mt = tt / 3, nt = tt - mt*3;
      int o = nt*16 + col;
      float sc = S1[l*48 + o], sh = H1[l*48 + o];
      for (int i = 0; i < 8; ++i) {
        int s = mt*16 + rbase + i;
        if (s < 36) {
          float v = fmaxf(acc[slot][i]*sc + sh, 0.f);
          sm_fpad[(s+4)*FS + l*48 + o] = (_Float16)v;
        }
      }
    }
  }

  // ---- phase 2: catconv (K padded 144 -> 160 with zeros) ----
  __syncthreads();
  _Float16* sm_w2 = sm_xin;                 // reuse (needs 64*168 <= 3*56*72)
  for (int i = tid; i < 64*FS; i += 256) sm_w2[i] = (_Float16)0.f;
  v8f z2 = {};
  v8f acc2[2]; acc2[0] = z2; acc2[1] = z2;
  for (int t = 0; t < 9; ++t) {
    __syncthreads();
    const _Float16* wsrc = W2H + (size_t)t*64*144;
    // async-stage 64x144 f16 slice: 64 rows * 18 x 16B chunks = 1152 (wave-aligned)
    for (int ch = tid; ch < 1152; ch += 256) {
      int o = ch / 18, part = ch - o*18;
      async_copy16(&sm_w2[o*FS + part*8], wsrc + o*144 + part*8);
    }
    if (t < 8) __builtin_prefetch(wsrc + 64*144, 0, 1);
    async_wait0();
    __syncthreads();
    for (int tt = wave, slot = 0; tt < 12; tt += 8, ++slot) {
      int mt = tt >> 2, nt = tt & 3;
      for (int kc = 0; kc < 5; ++kc) {
        v16h a = frag_ld(sm_fpad + (mt*16 + t)*FS + kc*32, FS);
        v16h b = frag_ld(sm_w2 + nt*16*FS + kc*32, FS);
        acc2[slot] = WMMA16(a, b, acc2[slot]);
      }
    }
  }
  for (int tt = wave, slot = 0; tt < 12; tt += 8, ++slot) {
    int mt = tt >> 2, nt = tt & 3;
    int c = nt*16 + col;
    float sc = S2[c], sh = H2[c];
    for (int i = 0; i < 8; ++i) {
      int s = mt*16 + rbase + i;
      if (s < 36) {
        float v = fmaxf(acc2[slot][i]*sc + sh, 0.f);
        roi_out[(size_t)n*2304 + c*36 + s] = (_Float16)v;   // k = c*36+s layout
      }
    }
  }
}

// ======================== kernel B: fc GEMM + LayerNorm + query ========================
__global__ __launch_bounds__(128) void fc_ln_kernel(
    const _Float16* __restrict__ roi, const _Float16* __restrict__ FCH,
    const float* __restrict__ fc_b, const float* __restrict__ ln_g,
    const float* __restrict__ ln_b, const float* __restrict__ fq_w,
    const float* __restrict__ fq_b, float* __restrict__ r_ws,
    _Float16* __restrict__ q_ws) {
  __shared__ float rbuf[16][68];
  __shared__ float smu[16], sinv[16];
  const int tid = threadIdx.x;
  const int wave = __builtin_amdgcn_readfirstlane(tid >> 5);
  const int lane = tid & 31;
  const int col = lane & 15, rbase = (lane >> 4) * 8, hi8 = (lane >> 4) << 3;
  const int n0 = blockIdx.x * 16;
  v8f acc = {};
  const _Float16* aBase = roi + (size_t)(n0 + col)*2304 + hi8;
  const _Float16* bBase = FCH + (size_t)(wave*16 + col)*2304 + hi8;
  for (int kc = 0; kc < 72; ++kc) {
    Frag a, b;
    a.h[0] = *(const v8h*)(aBase + kc*32);      a.h[1] = *(const v8h*)(aBase + kc*32 + 16);
    b.h[0] = *(const v8h*)(bBase + kc*32);      b.h[1] = *(const v8h*)(bBase + kc*32 + 16);
    acc = WMMA16(a.v, b.v, acc);
  }
  int c = wave*16 + col;
  float bias = fc_b[c];
  for (int i = 0; i < 8; ++i) rbuf[rbase + i][c] = acc[i] + bias;
  __syncthreads();
  if (tid < 16) {
    float m = 0.f;
    for (int j = 0; j < 64; ++j) m += rbuf[tid][j];
    m *= (1.f/64.f);
    float v = 0.f;
    for (int j = 0; j < 64; ++j) { float d = rbuf[tid][j] - m; v += d*d; }
    smu[tid] = m; sinv[tid] = rsqrtf(v*(1.f/64.f) + 1e-5f);
  }
  __syncthreads();
  for (int i = tid; i < 1024; i += 128) {
    int row = i >> 6, cc = i & 63;
    int nn = n0 + row;
    float val = (rbuf[row][cc] - smu[row]) * sinv[row] * ln_g[cc] + ln_b[cc];
    val = fmaxf(val, 0.f);
    r_ws[(size_t)nn*64 + cc] = val;
    int p = nn % 192;
    float q = fmaxf(val * fq_w[p] + fq_b[p], 0.f);
    q_ws[(size_t)nn*64 + cc] = (_Float16)q;
  }
}

// ======================== kernel C1: sampled depthwise attention ========================
__global__ __launch_bounds__(128) void attn_sample_kernel(
    const float* __restrict__ x,
    const float* __restrict__ a0_w,  const float* __restrict__ a0_b,
    const float* __restrict__ a01_w, const float* __restrict__ a01_b,
    const float* __restrict__ a11_w, const float* __restrict__ a11_b,
    const float* __restrict__ a21_w, const float* __restrict__ a21_b,
    _Float16* __restrict__ attn_s, _Float16* __restrict__ x_s) {
  __shared__ float xr[41][104];
  __shared__ float a0r[10][120];
  __shared__ float w0[25], w01[7], w11[11], w21[21];
  const int bb = blockIdx.x >> 6;
  const int c  = blockIdx.x & 63;
  const int tid = threadIdx.x;
  if (tid < 25)                    w0[tid]      = a0_w[c*25 + tid];
  else if (tid < 32)               w01[tid-25]  = a01_w[c*7  + tid-25];
  else if (tid >= 32 && tid < 43)  w11[tid-32]  = a11_w[c*11 + tid-32];
  else if (tid >= 43 && tid < 64)  w21[tid-43]  = a21_w[c*21 + tid-43];
  for (int i = tid; i < 10*120; i += 128) (&a0r[0][0])[i] = 0.f;
  const float* xc = x + (size_t)(bb*64 + c)*40*100;
  for (int i = tid; i < 41*104; i += 128) {
    int rr = i / 104, ww = i - rr*104;
    int r = rr - 2, w = ww - 2;
    xr[rr][ww] = (r >= 0 && r < 40 && w >= 0 && w < 100) ? xc[r*100 + w] : 0.f;
  }
  __syncthreads();
  float b0 = a0_b[c];
  for (int i = tid; i < 1000; i += 128) {        // a0: 5x5 dw at the 10 sampled rows
    int hi = i / 100, w = i - hi*100;
    int h = hi * 4;
    float acc = b0;
    for (int kh = 0; kh < 5; ++kh)
      for (int kw = 0; kw < 5; ++kw)
        acc += w0[kh*5 + kw] * xr[h + kh][w + kw];
    a0r[hi][w + 10] = acc;
  }
  __syncthreads();
  float b01 = a01_b[c], b11 = a11_b[c], b21 = a21_b[c];
  for (int i = tid; i < 250; i += 128) {
    int hi = i / 25, wi = i - hi*25;
    int wp = wi * 4;
    const float* row = a0r[hi];
    float att = row[wp + 10] + b01 + b11 + b21;
    for (int j = 0; j < 7;  ++j) att += w01[j] * row[wp + 7 + j];   // 1x7 pad3
    for (int j = 0; j < 11; ++j) att += w11[j] * row[wp + 5 + j];   // 1x11 pad5
    for (int j = 0; j < 21; ++j) att += w21[j] * row[wp + j];       // 1x21 pad10
    size_t base = ((size_t)bb*256 + i)*64 + c;
    attn_s[base] = (_Float16)att;
    x_s[base]    = (_Float16)xr[hi*4 + 2][wp + 2];
  }
  if (tid < 6) {                                 // zero pad rows 250..255
    size_t base = ((size_t)bb*256 + 250 + tid)*64 + c;
    attn_s[base] = (_Float16)0.f;
    x_s[base]    = (_Float16)0.f;
  }
}

// ======================== kernel C2: key/value 1x1 GEMMs ========================
__global__ __launch_bounds__(256) void kv_kernel(
    const _Float16* __restrict__ attn_s, const _Float16* __restrict__ x_s,
    const _Float16* __restrict__ FKA3H, const float* __restrict__ fka3b,
    const float* __restrict__ SK, const float* __restrict__ HK,
    const _Float16* __restrict__ FVH, const float* __restrict__ fval_b,
    _Float16* __restrict__ k_ws, _Float16* __restrict__ valT) {
  const int bb = blockIdx.x;
  const int tid = threadIdx.x;
  const int wave = __builtin_amdgcn_readfirstlane(tid >> 5);
  const int lane = tid & 31;
  const int col = lane & 15, rbase = (lane >> 4) * 8;
  const _Float16* A_att = attn_s + (size_t)bb*256*64;
  const _Float16* B_xs  = x_s    + (size_t)bb*256*64;
  // key^T[l][c] = relu(bn(FKA3 @ attn + FK@a3_b))
  for (int tile = wave; tile < 64; tile += 8) {
    int mt = tile >> 2, nt = tile & 3;
    v8f acc = {};
    for (int kc = 0; kc < 2; ++kc) {
      v16h a = frag_ld(A_att + (size_t)(mt*16)*64 + kc*32, 64);
      v16h b = frag_ld(FKA3H + (size_t)(nt*16)*64 + kc*32, 64);
      acc = WMMA16(a, b, acc);
    }
    int c = nt*16 + col;
    float bias = fka3b[c], sc = SK[c], sh = HK[c];
    for (int i = 0; i < 8; ++i) {
      int l = mt*16 + rbase + i;
      float v = fmaxf((acc[i] + bias)*sc + sh, 0.f);
      k_ws[(size_t)bb*16384 + l*64 + c] = (_Float16)v;
    }
  }
  // value^T[c][l] = fval_w @ x_s + fval_b
  for (int tile = wave; tile < 64; tile += 8) {
    int mt = tile >> 4, nt = tile & 15;
    v8f acc = {};
    for (int kc = 0; kc < 2; ++kc) {
      v16h a = frag_ld(FVH  + (size_t)(mt*16)*64 + kc*32, 64);
      v16h b = frag_ld(B_xs + (size_t)(nt*16)*64 + kc*32, 64);
      acc = WMMA16(a, b, acc);
    }
    for (int i = 0; i < 8; ++i) {
      int cr = mt*16 + rbase + i;
      float v = acc[i] + fval_b[cr];
      valT[(size_t)bb*16384 + cr*256 + nt*16 + col] = (_Float16)v;
    }
  }
}

// ======================== kernel D: attention ========================
__global__ __launch_bounds__(128) void attend_kernel(
    const _Float16* __restrict__ q_ws, const _Float16* __restrict__ k_ws,
    const _Float16* __restrict__ valT, const float* __restrict__ r_ws,
    const float* __restrict__ Ww, const float* __restrict__ Wb,
    float* __restrict__ out) {
  __shared__ float sS[16][264];
  __shared__ __align__(16) _Float16 sS16[16][264];
  const int bb = blockIdx.x / 12;
  const int mt = blockIdx.x - bb*12;
  const int tid = threadIdx.x;
  const int wave = __builtin_amdgcn_readfirstlane(tid >> 5);
  const int lane = tid & 31;
  const int col = lane & 15, rbase = (lane >> 4) * 8;
  const _Float16* Q = q_ws + ((size_t)bb*192 + mt*16)*64;
  const _Float16* K = k_ws + (size_t)bb*16384;
  const _Float16* V = valT + (size_t)bb*16384;
  // S = (Q @ K) * C^-0.5
  for (int nt = wave; nt < 16; nt += 4) {
    v8f acc = {};
    for (int kc = 0; kc < 2; ++kc) {
      v16h a = frag_ld(Q + kc*32, 64);
      v16h b = frag_ld(K + (size_t)(nt*16)*64 + kc*32, 64);
      acc = WMMA16(a, b, acc);
    }
    for (int i = 0; i < 8; ++i)
      sS[rbase + i][nt*16 + col] = acc[i] * 0.125f;
  }
  __syncthreads();
  if (tid < 16) {                      // row softmax over l < 250, pad with 0
    float m = -1e30f;
    for (int l = 0; l < 250; ++l) m = fmaxf(m, sS[tid][l]);
    float sum = 0.f;
    for (int l = 0; l < 250; ++l) { float e = __expf(sS[tid][l] - m); sS[tid][l] = e; sum += e; }
    float inv = 1.f / sum;
    for (int l = 0; l < 250; ++l) sS16[tid][l] = (_Float16)(sS[tid][l] * inv);
    for (int l = 250; l < 264; ++l) sS16[tid][l] = (_Float16)0.f;
  }
  __syncthreads();
  {                                    // ctx = S @ V ; out = r + ctx*Ww + Wb
    int nt = wave;                     // c-tile 0..3
    v8f acc = {};
    for (int kc = 0; kc < 8; ++kc) {
      v16h a = frag_ld(&sS16[0][0] + kc*32, 264);
      v16h b = frag_ld(V + (size_t)(nt*16)*256 + kc*32, 256);
      acc = WMMA16(a, b, acc);
    }
    int c = nt*16 + col;
    for (int i = 0; i < 8; ++i) {
      int p = mt*16 + rbase + i;
      int n = bb*192 + p;
      out[(size_t)n*64 + c] = r_ws[(size_t)n*64 + c] + acc[i]*Ww[p] + Wb[p];
    }
  }
}

// ======================== launcher ========================
extern "C" void kernel_launch(void* const* d_in, const int* in_sizes, int n_in,
                              void* d_out, int out_size, void* d_ws, size_t ws_size,
                              hipStream_t stream) {
  (void)in_sizes; (void)n_in; (void)out_size; (void)ws_size;
  const float* rf      = (const float*)d_in[0];
  const float* x       = (const float*)d_in[1];
  const float* convs_w = (const float*)d_in[2];
  const float* convs_g = (const float*)d_in[3];
  const float* convs_b = (const float*)d_in[4];
  const float* convs_m = (const float*)d_in[5];
  const float* convs_v = (const float*)d_in[6];
  const float* cat_w   = (const float*)d_in[7];
  const float* cat_g   = (const float*)d_in[8];
  const float* cat_b   = (const float*)d_in[9];
  const float* cat_m   = (const float*)d_in[10];
  const float* cat_v   = (const float*)d_in[11];
  const float* a0_w    = (const float*)d_in[12];
  const float* a0_b    = (const float*)d_in[13];
  const float* a01_w   = (const float*)d_in[14];
  const float* a01_b   = (const float*)d_in[15];
  const float* a11_w   = (const float*)d_in[16];
  const float* a11_b   = (const float*)d_in[17];
  const float* a21_w   = (const float*)d_in[18];
  const float* a21_b   = (const float*)d_in[19];
  const float* a3_w    = (const float*)d_in[20];
  const float* a3_b    = (const float*)d_in[21];
  const float* fkey_w  = (const float*)d_in[22];
  const float* fkey_g  = (const float*)d_in[23];
  const float* fkey_bb = (const float*)d_in[24];
  const float* fkey_m  = (const float*)d_in[25];
  const float* fkey_v  = (const float*)d_in[26];
  const float* fval_w  = (const float*)d_in[27];
  const float* fval_b  = (const float*)d_in[28];
  const float* fq_w    = (const float*)d_in[29];
  const float* fq_b    = (const float*)d_in[30];
  const float* Ww      = (const float*)d_in[31];
  const float* Wb      = (const float*)d_in[32];
  const float* fc_w    = (const float*)d_in[33];
  const float* fc_b    = (const float*)d_in[34];
  const float* ln_g    = (const float*)d_in[35];
  const float* ln_b    = (const float*)d_in[36];
  // d_in[37] = layer_index (unused by the reference forward)

  char* ws = (char*)d_ws;
  _Float16* W1H   = (_Float16*)(ws + OFF_W1H);
  _Float16* W2H   = (_Float16*)(ws + OFF_W2H);
  _Float16* FCH   = (_Float16*)(ws + OFF_FCH);
  _Float16* FVH   = (_Float16*)(ws + OFF_FVH);
  _Float16* FKA3H = (_Float16*)(ws + OFF_FKA3H);
  float*    FKA3B = (float*)(ws + OFF_FKA3B);
  float*    S1    = (float*)(ws + OFF_S1);
  float*    H1    = (float*)(ws + OFF_H1);
  float*    S2    = (float*)(ws + OFF_S2);
  float*    H2    = (float*)(ws + OFF_H2);
  float*    SK    = (float*)(ws + OFF_SK);
  float*    HK    = (float*)(ws + OFF_HK);
  _Float16* ROI   = (_Float16*)(ws + OFF_ROI);
  float*    RWS   = (float*)(ws + OFF_RWS);
  _Float16* QWS   = (_Float16*)(ws + OFF_QWS);
  _Float16* ATT   = (_Float16*)(ws + OFF_ATT);
  _Float16* XSM   = (_Float16*)(ws + OFF_XSM);
  _Float16* KWS   = (_Float16*)(ws + OFF_KWS);
  _Float16* VT    = (_Float16*)(ws + OFF_VT);
  float*    out   = (float*)d_out;

  // prep
  prep_w1<<<(3*9*48*64 + 255)/256, 256, 0, stream>>>(convs_w, W1H);
  prep_w2<<<(9*64*144 + 255)/256, 256, 0, stream>>>(cat_w, W2H);
  prep_cast_f16<<<(64*2304 + 255)/256, 256, 0, stream>>>(fc_w, FCH, 64*2304);
  prep_cast_f16<<<(64*64 + 255)/256, 256, 0, stream>>>(fval_w, FVH, 64*64);
  prep_fka3<<<16, 256, 0, stream>>>(fkey_w, a3_w, a3_b, FKA3H, FKA3B);
  prep_bn<<<1, 320, 0, stream>>>(convs_g, convs_b, convs_m, convs_v,
                                 cat_g, cat_b, cat_m, cat_v,
                                 fkey_g, fkey_bb, fkey_m, fkey_v,
                                 S1, H1, S2, H2, SK, HK);
  // heavy path
  roi_conv_kernel<<<3072, 256, 0, stream>>>(rf, W1H, W2H, S1, H1, S2, H2, ROI);
  fc_ln_kernel<<<192, 128, 0, stream>>>(ROI, FCH, fc_b, ln_g, ln_b, fq_w, fq_b, RWS, QWS);
  // attention path
  attn_sample_kernel<<<16*64, 128, 0, stream>>>(x, a0_w, a0_b, a01_w, a01_b,
                                                a11_w, a11_b, a21_w, a21_b, ATT, XSM);
  kv_kernel<<<16, 256, 0, stream>>>(ATT, XSM, FKA3H, FKA3B, SK, HK, FVH, fval_b, KWS, VT);
  attend_kernel<<<192, 128, 0, stream>>>(QWS, KWS, VT, RWS, Ww, Wb, out);
}
